// LGFFN_84387517432405
// MI455X (gfx1250) — matrix operand order repaired and checked by reading, not compiled, not run on hardware
//
#include <hip/hip_runtime.h>
#include <cstdint>
#include <cstddef>

// ---------------------------------------------------------------------------
// LGFFN pyramid aggregator for MI455X (gfx1250, wave32, WMMA)
// Pointwise convs -> v_wmma_f32_16x16x32_f16 GEMMs with fused epilogues,
// double-buffered async global->LDS tile staging (ASYNCcnt pipeline).
// ---------------------------------------------------------------------------

#define WB  4      // batch
#define CCH 256    // channels

typedef _Float16 h16;
typedef h16  v16h __attribute__((ext_vector_type(16)));
typedef h16  v8h  __attribute__((ext_vector_type(8)));
typedef h16  v4h  __attribute__((ext_vector_type(4)));
typedef float v8f __attribute__((ext_vector_type(8)));

#if defined(__gfx1250__)
# if __has_builtin(__builtin_amdgcn_global_load_async_to_lds_b128) && \
     __has_builtin(__builtin_amdgcn_s_wait_asynccnt)
#  define USE_ASYNC_LDS 1
# endif
#endif
#ifndef USE_ASYNC_LDS
# define USE_ASYNC_LDS 0
#endif

#if USE_ASYNC_LDS
typedef int v4i __attribute__((ext_vector_type(4)));
typedef __attribute__((address_space(1))) v4i* gldp;   // global int4*
typedef __attribute__((address_space(3))) v4i* sldp;   // LDS int4*
#endif

// -------------------------- device helpers ---------------------------------

__device__ __forceinline__ float wave_sum32(float v) {
#pragma unroll
  for (int m = 16; m >= 1; m >>= 1) v += __shfl_xor(v, m, 32);
  return v;
}

__device__ __forceinline__ float gelu_exact(float x) {
  return 0.5f * x * (1.0f + erff(x * 0.70710678118654752f));
}

// -------------------------- small utility kernels --------------------------

__global__ void k_zero(float* __restrict__ p, int n) {
  int i = blockIdx.x * blockDim.x + threadIdx.x;
  if (i < n) p[i] = 0.f;
}

// fp32 -> f16 (weights)
__global__ void k_cvt(const float* __restrict__ src, h16* __restrict__ dst, int n) {
  int i = (blockIdx.x * blockDim.x + threadIdx.x) * 4;
  if (i + 3 < n) {
    float4 f = *(const float4*)(src + i);
    v4h o;
    o[0] = (h16)f.x; o[1] = (h16)f.y; o[2] = (h16)f.z; o[3] = (h16)f.w;
    *(v4h*)(dst + i) = o;
  }
}

// -------------------------- LayerNorm / conv prep kernels ------------------
// One wave32 per spatial position; each lane owns 8 contiguous channels.

// t[pos, c] = f16( LN_n(x)[b,c,p] )     (LocalAggregator prep)
__global__ void k_la_prep(const float* __restrict__ x,
                          const float* __restrict__ ng, const float* __restrict__ nb,
                          h16* __restrict__ t, int HW) {
  int wave = threadIdx.x >> 5, lane = threadIdx.x & 31;
  int pos = blockIdx.x * 8 + wave;
  int b = pos / HW, p = pos % HW;
  const float* xp = x + ((size_t)b * CCH) * HW + p;
  float v[8]; float s = 0.f;
#pragma unroll
  for (int i = 0; i < 8; ++i) { v[i] = xp[(size_t)(lane * 8 + i) * HW]; s += v[i]; }
  float mean = wave_sum32(s) * (1.f / CCH);
  float q = 0.f;
#pragma unroll
  for (int i = 0; i < 8; ++i) { v[i] -= mean; q += v[i] * v[i]; }
  float rs = rsqrtf(wave_sum32(q) * (1.f / CCH) + 1e-6f);
  v8h o;
#pragma unroll
  for (int i = 0; i < 8; ++i) {
    int c = lane * 8 + i;
    o[i] = (h16)(v[i] * rs * ng[c] + nb[c]);
  }
  *(v8h*)(t + (size_t)pos * CCH + lane * 8) = o;
}

// t[pos, c] = f16( LN_n(dw3(x, stride1))[b,c,p] )   (GlobalAggregator prep)
__global__ void k_ga_prep(const float* __restrict__ x, const float* __restrict__ w,
                          const float* __restrict__ ng, const float* __restrict__ nb,
                          h16* __restrict__ t, int H, int W) {
  int HW = H * W;
  int wave = threadIdx.x >> 5, lane = threadIdx.x & 31;
  int pos = blockIdx.x * 8 + wave;
  int b = pos / HW, p = pos % HW;
  int y = p / W, xx = p % W;
  float v[8]; float s = 0.f;
#pragma unroll
  for (int i = 0; i < 8; ++i) {
    int c = lane * 8 + i;
    const float* xc = x + ((size_t)(b * CCH + c)) * HW;
    const float* wc = w + c * 9;
    float a = 0.f;
#pragma unroll
    for (int ky = 0; ky < 3; ++ky) {
      int yy = y + ky - 1;
      if ((unsigned)yy < (unsigned)H) {
#pragma unroll
        for (int kx = 0; kx < 3; ++kx) {
          int xc2 = xx + kx - 1;
          if ((unsigned)xc2 < (unsigned)W) a += xc[yy * W + xc2] * wc[ky * 3 + kx];
        }
      }
    }
    v[i] = a; s += a;
  }
  float mean = wave_sum32(s) * (1.f / CCH);
  float q = 0.f;
#pragma unroll
  for (int i = 0; i < 8; ++i) { v[i] -= mean; q += v[i] * v[i]; }
  float rs = rsqrtf(wave_sum32(q) * (1.f / CCH) + 1e-6f);
  v8h o;
#pragma unroll
  for (int i = 0; i < 8; ++i) {
    int c = lane * 8 + i;
    o[i] = (h16)(v[i] * rs * ng[c] + nb[c]);
  }
  *(v8h*)(t + (size_t)pos * CCH + lane * 8) = o;
}

// u = LN_nu(dw3(x, stride2))  (fp32, NCHW at half resolution)
__global__ void k_down(const float* __restrict__ x, const float* __restrict__ w,
                       const float* __restrict__ nug, const float* __restrict__ nub,
                       float* __restrict__ u, int H, int W) {
  int Ho = H >> 1, Wo = W >> 1;
  int HWo = Ho * Wo, HW = H * W;
  int wave = threadIdx.x >> 5, lane = threadIdx.x & 31;
  int pos = blockIdx.x * 8 + wave;
  int b = pos / HWo, p = pos % HWo;
  int yo = p / Wo, xo = p % Wo;
  float v[8]; float s = 0.f;
#pragma unroll
  for (int i = 0; i < 8; ++i) {
    int c = lane * 8 + i;
    const float* xc = x + ((size_t)(b * CCH + c)) * HW;
    const float* wc = w + c * 9;
    float a = 0.f;
#pragma unroll
    for (int ky = 0; ky < 3; ++ky) {
      int yy = 2 * yo + ky - 1;
      if ((unsigned)yy < (unsigned)H) {
#pragma unroll
        for (int kx = 0; kx < 3; ++kx) {
          int xx2 = 2 * xo + kx - 1;
          if ((unsigned)xx2 < (unsigned)W) a += xc[yy * W + xx2] * wc[ky * 3 + kx];
        }
      }
    }
    v[i] = a; s += a;
  }
  float mean = wave_sum32(s) * (1.f / CCH);
  float q = 0.f;
#pragma unroll
  for (int i = 0; i < 8; ++i) { v[i] -= mean; q += v[i] * v[i]; }
  float rs = rsqrtf(wave_sum32(q) * (1.f / CCH) + 1e-6f);
#pragma unroll
  for (int i = 0; i < 8; ++i) {
    int c = lane * 8 + i;
    u[((size_t)(b * CCH + c)) * HWo + p] = v[i] * rs * nug[c] + nub[c];
  }
}

// u = up2( LN_nu(x) )  depthwise ConvTranspose2d k=2 s=2 (fp32 NCHW, 2H x 2W)
__global__ void k_up(const float* __restrict__ x, const float* __restrict__ upw,
                     const float* __restrict__ upb, const float* __restrict__ nug,
                     const float* __restrict__ nub, float* __restrict__ u, int H, int W) {
  int HW = H * W;
  int wave = threadIdx.x >> 5, lane = threadIdx.x & 31;
  int pos = blockIdx.x * 8 + wave;
  int b = pos / HW, p = pos % HW;
  int y = p / W, xx = p % W;
  const float* xp = x + ((size_t)b * CCH) * HW + p;
  float v[8]; float s = 0.f;
#pragma unroll
  for (int i = 0; i < 8; ++i) { v[i] = xp[(size_t)(lane * 8 + i) * HW]; s += v[i]; }
  float mean = wave_sum32(s) * (1.f / CCH);
  float q = 0.f;
#pragma unroll
  for (int i = 0; i < 8; ++i) { v[i] -= mean; q += v[i] * v[i]; }
  float rs = rsqrtf(wave_sum32(q) * (1.f / CCH) + 1e-6f);
  int Wo = 2 * W;
  size_t HWo = (size_t)4 * HW;
#pragma unroll
  for (int i = 0; i < 8; ++i) {
    int c = lane * 8 + i;
    float ln = v[i] * rs * nug[c] + nub[c];
    float bb = upb[c];
    float* up = u + ((size_t)(b * CCH + c)) * HWo + (size_t)(2 * y) * Wo + 2 * xx;
    up[0]      = ln * upw[c * 4 + 0] + bb;
    up[1]      = ln * upw[c * 4 + 1] + bb;
    up[Wo]     = ln * upw[c * 4 + 2] + bb;
    up[Wo + 1] = ln * upw[c * 4 + 3] + bb;
  }
}

// GRN finalize: scale[b,ic] = 1 + grn_g[ic] * (Gx / (Gx + 1e-6)), Gx = sqrt(sumsq)
__global__ void k_grn_fin(const float* __restrict__ gsum, const float* __restrict__ gg,
                          float* __restrict__ scale, int IC) {
  int t = blockIdx.x * blockDim.x + threadIdx.x;
  if (t < WB * IC) {
    float g = sqrtf(gsum[t]);
    float nx = g / (g + 1e-6f);
    scale[t] = 1.f + gg[t % IC] * nx;
  }
}

// -------------------------- WMMA GEMM with fused epilogues -----------------
// D(M x N) = A(M x K, f16 row-major weights) * B(K x N) + epilogue.
// B is stored position-major: Bm[n*K + k] (activations).
// Macro-tile 128x128 per 256-thread block (8 waves); wave tile 32(M) x 64(N)
// -> 8 accumulators, 8 WMMA per K-step. Double-buffered LDS; tiles staged via
// GLOBAL_LOAD_ASYNC_TO_LDS_B128 (ASYNCcnt) and overlapped with WMMA.
// epi==0: h = gelu(acc + bias) -> hOut[n*M+m] (f16); optional GRN sum-of-squares
// epi==1: s = sigmoid(acc + bias); out = (1-a)*x*s + a*x  (store or accumulate)

__launch_bounds__(256)
__global__ void k_gemm(const h16* __restrict__ A, const h16* __restrict__ Bm,
                       const float* __restrict__ bias,
                       int M, int N, int K, int HW, int epi,
                       h16* __restrict__ hOut, float* __restrict__ gsum,
                       const float* __restrict__ bscale, const float* __restrict__ bshift,
                       const float* __restrict__ xres, const float* __restrict__ alphap,
                       float* __restrict__ dest, int accum) {
  __shared__ __align__(16) h16 As[2][128 * 32];
  __shared__ __align__(16) h16 Bs[2][128 * 32];

  const int tid  = threadIdx.x;
  const int lane = tid & 31;
  const int wave = tid >> 5;
  const int m0 = blockIdx.y * 128;
  const int n0 = blockIdx.x * 128;
  const int mt = wave & 3;        // wave's 32-row M sub-tile
  const int ng = wave >> 2;       // wave's 64-col N group
  const int l15 = lane & 15;

  v8f acc[8] = {};

  // --- tile staging: 128x32 A and 128x32 B, 2x v8h chunks per thread each
  auto stage = [&](int k0, int buf) {
    h16* dA = &As[buf][0];
    h16* dB = &Bs[buf][0];
    if (!bscale) {
#pragma unroll
      for (int i = 0; i < 2; ++i) {
        int idx = tid + i * 256;
        int row = idx >> 2, ch = (idx & 3) * 8;
        const h16* gA = A  + (size_t)(m0 + row) * K + k0 + ch;
        const h16* gB = Bm + (size_t)(n0 + row) * K + k0 + ch;
#if USE_ASYNC_LDS
        __builtin_amdgcn_global_load_async_to_lds_b128(
            (gldp)gA, (sldp)(dA + row * 32 + ch), 0, 0);
        __builtin_amdgcn_global_load_async_to_lds_b128(
            (gldp)gB, (sldp)(dB + row * 32 + ch), 0, 0);
#else
        *(v8h*)(dA + row * 32 + ch) = *(const v8h*)gA;
        *(v8h*)(dB + row * 32 + ch) = *(const v8h*)gB;
#endif
      }
    } else {
      // synchronous path with the GRN affine folded into the B tile
#pragma unroll
      for (int i = 0; i < 2; ++i) {
        int idx = tid + i * 256;
        int row = idx >> 2, ch = (idx & 3) * 8;
        *(v8h*)(dA + row * 32 + ch) =
            *(const v8h*)(A + (size_t)(m0 + row) * K + k0 + ch);
        v8h v = *(const v8h*)(Bm + (size_t)(n0 + row) * K + k0 + ch);
        int bb = (n0 + row) / HW;
#pragma unroll
        for (int r = 0; r < 8; ++r) {
          int kk = k0 + ch + r;
          float f = (float)v[r];
          f = f * bscale[bb * K + kk] + bshift[kk];
          v[r] = (h16)f;
        }
        *(v8h*)(dB + row * 32 + ch) = v;
      }
    }
  };

  stage(0, 0);

  for (int k0 = 0, it = 0; k0 < K; k0 += 32, ++it) {
    const int buf = it & 1;
#if USE_ASYNC_LDS
    if (!bscale) __builtin_amdgcn_s_wait_asynccnt(0);
#endif
    __syncthreads();                    // tile `it` visible; prev compute done
    if (k0 + 32 < K) stage(k0 + 32, buf ^ 1);   // DMA next tile || compute

    const h16* sA = &As[buf][0];
    const h16* sB = &Bs[buf][0];
    // A fragments (16x32): lanes 0-15 hold K 0-7 & 16-23; lanes 16-31 K 8-15 & 24-31
    union FRAG { v16h v; v8h h[2]; };
    FRAG af0, af1;
    const int ak = (lane < 16) ? 0 : 8;
    af0.h[0] = *(const v8h*)(sA + (mt * 32 + l15) * 32 + ak);
    af0.h[1] = *(const v8h*)(sA + (mt * 32 + l15) * 32 + ak + 16);
    af1.h[0] = *(const v8h*)(sA + (mt * 32 + 16 + l15) * 32 + ak);
    af1.h[1] = *(const v8h*)(sA + (mt * 32 + 16 + l15) * 32 + ak + 16);
    const int bk = (lane < 16) ? 0 : 8;
#pragma unroll
    for (int j = 0; j < 4; ++j) {
      FRAG bf;
      int bn = ng * 64 + j * 16 + l15;
      bf.h[0] = *(const v8h*)(sB + bn * 32 + bk);
      bf.h[1] = *(const v8h*)(sB + bn * 32 + bk + 16);
      acc[j] = __builtin_amdgcn_wmma_f32_16x16x32_f16(
          false, af0.v, false, bf.v, (short)0, acc[j], false, false);
      acc[4 + j] = __builtin_amdgcn_wmma_f32_16x16x32_f16(
          false, af1.v, false, bf.v, (short)0, acc[4 + j], false, false);
    }
  }

  // --- epilogue.  D layout: lanes 0-15: VGPR r -> (M=m16+r, N=l15);
  //                lanes 16-31: VGPR r -> (M=m16+8+r, N=l15)
  if (epi == 0) {
#pragma unroll
    for (int ms = 0; ms < 2; ++ms) {
      const int mbase = m0 + mt * 32 + ms * 16 + ((lane < 16) ? 0 : 8);
#pragma unroll
      for (int j = 0; j < 4; ++j) {
        int n = n0 + ng * 64 + j * 16 + l15;
        int bb = n / HW;
        float sq[8];
#pragma unroll
        for (int r = 0; r < 8; ++r) {
          int m = mbase + r;
          float g = gelu_exact(acc[ms * 4 + j][r] + bias[m]);
          hOut[(size_t)n * M + m] = (h16)g;
          sq[r] = g * g;
        }
        if (gsum) {
#pragma unroll
          for (int r = 0; r < 8; ++r) {
            float red = sq[r];
#pragma unroll
            for (int msk = 8; msk >= 1; msk >>= 1) red += __shfl_xor(red, msk, 32);
            if (l15 == 0) atomicAdd(&gsum[bb * M + mbase + r], red);
          }
        }
      }
    }
  } else {
    float a = *alphap;
#pragma unroll
    for (int ms = 0; ms < 2; ++ms) {
      const int mbase = m0 + mt * 32 + ms * 16 + ((lane < 16) ? 0 : 8);
#pragma unroll
      for (int j = 0; j < 4; ++j) {
        int n = n0 + ng * 64 + j * 16 + l15;
        int bb = n / HW, p = n - bb * HW;
#pragma unroll
        for (int r = 0; r < 8; ++r) {
          int m = mbase + r;
          float v = acc[ms * 4 + j][r] + bias[m];
          float s = 1.f / (1.f + __expf(-v));
          size_t idx = ((size_t)(bb * M + m)) * HW + p;   // M == CCH here
          float xv = xres[idx];
          float o = (1.f - a) * xv * s + a * xv;
          if (accum) dest[idx] += o; else dest[idx] = o;
        }
      }
    }
  }
}

// -------------------------- host-side orchestration ------------------------

struct GAp {
  const float *alpha, *b1, *b2, *down_w, *dw_w, *grn_b, *grn_g,
              *n_b, *n_g, *nu_b, *nu_g, *w1, *w2;
  bool down;
};
struct LAp {
  const float *alpha, *b1, *b2, *n_b, *n_g, *nu_b, *nu_g,
              *up_b, *up_w, *w1, *w2;
  bool up;
};

// workspace layout (bytes)
static const size_t OFF_W1  = 0;
static const size_t OFF_W2  = OFF_W1 + ((size_t)1 << 20);
static const size_t OFF_T   = OFF_W2 + ((size_t)1 << 20);          // f16, <= 32 MB
static const size_t OFF_H   = OFF_T  + ((size_t)32 << 20);         // f16, <= 128 MB
static const size_t OFF_X   = OFF_H  + ((size_t)128 << 20);        // fp32, <= 64 MB
static const size_t OFF_GS  = OFF_X  + ((size_t)64 << 20);
static const size_t OFF_SC  = OFF_GS + ((size_t)64 << 10);
static const size_t OFF_P31 = OFF_SC + ((size_t)64 << 10);         // fp32, 4 MB
static const size_t OFF_P21 = OFF_P31 + ((size_t)4 << 20);         // fp32, 16 MB

static void run_ga(const GAp& g, const float* x, int H, float* dest, int accum,
                   char* ws, hipStream_t st) {
  const int IC = 4 * CCH;
  int He = g.down ? (H >> 1) : H;
  int HWe = He * He;
  int Np = WB * HWe;
  h16* W1h = (h16*)(ws + OFF_W1);
  h16* W2h = (h16*)(ws + OFF_W2);
  h16* t   = (h16*)(ws + OFF_T);
  h16* h   = (h16*)(ws + OFF_H);
  float* xr = (float*)(ws + OFF_X);
  float* gs = (float*)(ws + OFF_GS);
  float* sc = (float*)(ws + OFF_SC);
  const float* xe = x;
  if (g.down) {
    k_down<<<Np / 8, 256, 0, st>>>(x, g.down_w, g.nu_g, g.nu_b, xr, H, H);
    xe = xr;
  }
  k_cvt<<<(IC * CCH / 4 + 255) / 256, 256, 0, st>>>(g.w1, W1h, IC * CCH);
  k_cvt<<<(CCH * IC / 4 + 255) / 256, 256, 0, st>>>(g.w2, W2h, CCH * IC);
  k_ga_prep<<<Np / 8, 256, 0, st>>>(xe, g.dw_w, g.n_g, g.n_b, t, He, He);
  k_zero<<<(WB * IC + 255) / 256, 256, 0, st>>>(gs, WB * IC);
  k_gemm<<<dim3(Np / 128, IC / 128), 256, 0, st>>>(
      W1h, t, g.b1, IC, Np, CCH, HWe, 0,
      h, gs, nullptr, nullptr, nullptr, nullptr, nullptr, 0);
  k_grn_fin<<<(WB * IC + 255) / 256, 256, 0, st>>>(gs, g.grn_g, sc, IC);
  k_gemm<<<dim3(Np / 128, CCH / 128), 256, 0, st>>>(
      W2h, h, g.b2, CCH, Np, IC, HWe, 1,
      nullptr, nullptr, sc, g.grn_b, xe, g.alpha, dest, accum);
}

static void run_la(const LAp& l, const float* x, int H, float* dest, int accum,
                   char* ws, hipStream_t st) {
  const int IC = 2 * CCH;
  int He = l.up ? (2 * H) : H;
  int HWe = He * He;
  int Np = WB * HWe;
  h16* W1h = (h16*)(ws + OFF_W1);
  h16* W2h = (h16*)(ws + OFF_W2);
  h16* t   = (h16*)(ws + OFF_T);
  h16* h   = (h16*)(ws + OFF_H);
  float* xr = (float*)(ws + OFF_X);
  const float* xe = x;
  if (l.up) {
    int Npin = WB * H * H;
    k_up<<<Npin / 8, 256, 0, st>>>(x, l.up_w, l.up_b, l.nu_g, l.nu_b, xr, H, H);
    xe = xr;
  }
  k_cvt<<<(IC * CCH / 4 + 255) / 256, 256, 0, st>>>(l.w1, W1h, IC * CCH);
  k_cvt<<<(CCH * IC / 4 + 255) / 256, 256, 0, st>>>(l.w2, W2h, CCH * IC);
  k_la_prep<<<Np / 8, 256, 0, st>>>(xe, l.n_g, l.n_b, t, HWe);
  k_gemm<<<dim3(Np / 128, IC / 128), 256, 0, st>>>(
      W1h, t, l.b1, IC, Np, CCH, HWe, 0,
      h, nullptr, nullptr, nullptr, nullptr, nullptr, nullptr, 0);
  k_gemm<<<dim3(Np / 128, CCH / 128), 256, 0, st>>>(
      W2h, h, l.b2, CCH, Np, IC, HWe, 1,
      nullptr, nullptr, nullptr, nullptr, xe, l.alpha, dest, accum);
}

extern "C" void kernel_launch(void* const* d_in, const int* in_sizes, int n_in,
                              void* d_out, int out_size, void* d_ws, size_t ws_size,
                              hipStream_t stream) {
  (void)in_sizes; (void)n_in; (void)out_size; (void)ws_size;
  int gi = 0;
  auto nf = [&]() { return (const float*)d_in[gi++]; };
  const float* p1 = nf();
  const float* p2 = nf();
  const float* p3 = nf();
  const float* p4 = nf();

  // JAX flattens dicts in sorted-key order.
  auto read_ga = [&](bool down) {
    GAp g{}; g.down = down;
    g.alpha = nf(); g.b1 = nf(); g.b2 = nf();
    if (down) g.down_w = nf();
    g.dw_w = nf(); g.grn_b = nf(); g.grn_g = nf(); g.n_b = nf(); g.n_g = nf();
    if (down) { g.nu_b = nf(); g.nu_g = nf(); }
    g.w1 = nf(); g.w2 = nf();
    return g;
  };
  auto read_la = [&](bool up) {
    LAp l{}; l.up = up;
    l.alpha = nf(); l.b1 = nf(); l.b2 = nf(); l.n_b = nf(); l.n_g = nf();
    if (up) { l.nu_b = nf(); l.nu_g = nf(); l.up_b = nf(); l.up_w = nf(); }
    l.w1 = nf(); l.w2 = nf();
    return l;
  };

  GAp m_p12_gca  = read_ga(false);   // p12_gca_p10
  LAp m_p12_lca  = read_la(true);    // p12_lca_p21
  GAp m_p21_gca  = read_ga(false);   // p21_gca_p20
  LAp m_p21_lca  = read_la(true);    // p21_lca_p31
  GAp m_p22_gca  = read_ga(true);    // p22_gca_p12
  LAp m_p22_lcaP = read_la(false);   // p22_lca_P21
  LAp m_p22_lca  = read_la(false);   // p22_lca_p20
  GAp m_p31_gca  = read_ga(false);   // p31_gca_p30
  LAp m_p31_lca  = read_la(true);    // p31_lca_p40
  GAp m_p32_gca  = read_ga(true);    // p32_gca_p22
  LAp m_p32_lcaP = read_la(false);   // p32_lca_P31
  LAp m_p32_lca  = read_la(false);   // p32_lca_p30
  GAp m_p42_gca  = read_ga(true);    // p42_gca_p32
  LAp m_p42_lca  = read_la(false);   // p42_lca_p40

  char* ws = (char*)d_ws;
  float* out = (float*)d_out;
  float* o_p12 = out;
  float* o_p22 = o_p12 + (size_t)WB * CCH * 128 * 128;
  float* o_p32 = o_p22 + (size_t)WB * CCH * 64 * 64;
  float* o_p42 = o_p32 + (size_t)WB * CCH * 32 * 32;
  float* w_p31 = (float*)(ws + OFF_P31);
  float* w_p21 = (float*)(ws + OFF_P21);

  // p3_1 = GA(p3) + LA(p4, up)
  run_ga(m_p31_gca, p3, 32, w_p31, 0, ws, stream);
  run_la(m_p31_lca, p4, 16, w_p31, 1, ws, stream);
  // p2_1 = GA(p2) + LA(p3_1, up)
  run_ga(m_p21_gca, p2, 64, w_p21, 0, ws, stream);
  run_la(m_p21_lca, w_p31, 32, w_p21, 1, ws, stream);
  // p1_2 = GA(p1) + LA(p2_1, up)
  run_ga(m_p12_gca, p1, 128, o_p12, 0, ws, stream);
  run_la(m_p12_lca, w_p21, 64, o_p12, 1, ws, stream);
  // p2_2 = LA(p2) + LA(p2_1) + GA(p1_2, down)
  run_la(m_p22_lca,  p2,    64, o_p22, 0, ws, stream);
  run_la(m_p22_lcaP, w_p21, 64, o_p22, 1, ws, stream);
  run_ga(m_p22_gca,  o_p12, 128, o_p22, 1, ws, stream);
  // p3_2 = LA(p3) + LA(p3_1) + GA(p2_2, down)
  run_la(m_p32_lca,  p3,    32, o_p32, 0, ws, stream);
  run_la(m_p32_lcaP, w_p31, 32, o_p32, 1, ws, stream);
  run_ga(m_p32_gca,  o_p22, 64, o_p32, 1, ws, stream);
  // p4_2 = LA(p4) + GA(p3_2, down)
  run_la(m_p42_lca, p4,    16, o_p42, 0, ws, stream);
  run_ga(m_p42_gca, o_p32, 32, o_p42, 1, ws, stream);
}
